// RelationNetBasedAggregationFunction_67903432950387
// MI455X (gfx1250) — matrix operand order repaired and checked by reading, not compiled, not run on hardware
//
#include <hip/hip_runtime.h>
#include <hip/hip_bf16.h>
#include <math.h>
#include <stdint.h>

// ---------------------------------------------------------------------------
// Problem constants (from reference)
// ---------------------------------------------------------------------------
static constexpr int BATCH   = 8;
static constexpr int NNODES  = 2048;
static constexpr int F_IN    = 256;
static constexpr int F_MSG   = 128;
static constexpr int TOPK    = 32;
static constexpr int ROWS    = BATCH * NNODES;          // 16384 flattened nodes
static constexpr float BN_EPS      = 1e-5f;
static constexpr float LEAKY_SLOPE = 0.01f;

typedef __attribute__((ext_vector_type(16))) _Float16 v16h;
typedef __attribute__((ext_vector_type(8)))  float    v8f;

// ---------------------------------------------------------------------------
// Kernel 1: projection GEMM  Xm[r, c] = sum_k X[r,k] * W[c,k] + b[c]
// WMMA f32 = f16 x f16 (16x16x32), accumulate over K=256 (8 WMMA ops).
// Block: 256 threads = 8 waves; wave j computes columns [16j, 16j+16) of a
// 16-row tile. Grid.x = ROWS/16 = 1024.
// ---------------------------------------------------------------------------
__global__ __launch_bounds__(256)
void proj_gemm_wmma(const float* __restrict__ X,
                    const float* __restrict__ W,
                    const float* __restrict__ bias,
                    float* __restrict__ Xm) {
    const int lane  = threadIdx.x & 31;
    const int wave  = threadIdx.x >> 5;   // 0..7 -> column tile
    const int rtile = blockIdx.x;         // 0..1023 -> 16-row tile
    const int mloc  = lane & 15;
    const int khalf = lane >> 4;          // 0 or 1

    // A-matrix source: X row for this lane's M
    const float* xrow = X + (size_t)(rtile * 16 + mloc) * F_IN;
    // B-matrix source: W row == B column (since B[k,n] = W[n,k])
    const float* wrow = W + (size_t)(wave * 16 + mloc) * F_IN;

    const float4* xr4 = reinterpret_cast<const float4*>(xrow);
    const float4* wr4 = reinterpret_cast<const float4*>(wrow);

    v8f acc = {};
#pragma unroll
    for (int kk = 0; kk < F_IN / 32; ++kk) {
        const int k0 = kk * 32;

        // ---- A tile (16x32 f16): lane holds M=mloc, K in
        //      [k0+khalf*8, +8) -> v16h[0..7], [k0+khalf*8+16, +8) -> v16h[8..15]
        const int ka = k0 + khalf * 8;               // multiple of 8 -> f4-aligned
        float4 p0 = xr4[(ka >> 2) + 0];
        float4 p1 = xr4[(ka >> 2) + 1];
        float4 p2 = xr4[((ka + 16) >> 2) + 0];
        float4 p3 = xr4[((ka + 16) >> 2) + 1];
        v16h a;
        a[0]=(_Float16)p0.x; a[1]=(_Float16)p0.y; a[2]=(_Float16)p0.z; a[3]=(_Float16)p0.w;
        a[4]=(_Float16)p1.x; a[5]=(_Float16)p1.y; a[6]=(_Float16)p1.z; a[7]=(_Float16)p1.w;
        a[8]=(_Float16)p2.x; a[9]=(_Float16)p2.y; a[10]=(_Float16)p2.z; a[11]=(_Float16)p2.w;
        a[12]=(_Float16)p3.x; a[13]=(_Float16)p3.y; a[14]=(_Float16)p3.z; a[15]=(_Float16)p3.w;

        // ---- B tile (32x16 f16): lane holds N=mloc, K in
        //      [k0+khalf*8, +8) -> v16h[0..7], [k0+khalf*8+16, +8) -> v16h[8..15]
        const int kb = k0 + khalf * 8;               // multiple of 8 -> f4-aligned
        float4 q0 = wr4[(kb >> 2) + 0];
        float4 q1 = wr4[(kb >> 2) + 1];
        float4 q2 = wr4[((kb + 16) >> 2) + 0];
        float4 q3 = wr4[((kb + 16) >> 2) + 1];
        v16h bmat;
        bmat[0]=(_Float16)q0.x;  bmat[1]=(_Float16)q0.y;  bmat[2]=(_Float16)q0.z;  bmat[3]=(_Float16)q0.w;
        bmat[4]=(_Float16)q1.x;  bmat[5]=(_Float16)q1.y;  bmat[6]=(_Float16)q1.z;  bmat[7]=(_Float16)q1.w;
        bmat[8]=(_Float16)q2.x;  bmat[9]=(_Float16)q2.y;  bmat[10]=(_Float16)q2.z; bmat[11]=(_Float16)q2.w;
        bmat[12]=(_Float16)q3.x; bmat[13]=(_Float16)q3.y; bmat[14]=(_Float16)q3.z; bmat[15]=(_Float16)q3.w;

        // D = A x B + C
        acc = __builtin_amdgcn_wmma_f32_16x16x32_f16(
            /*neg_a=*/false, a, /*neg_b=*/false, bmat,
            /*c_mod=*/(short)0, acc, /*reuse_a=*/false, /*reuse_b=*/false);
    }

    // D layout: VGPR r, lane<16 -> row r col lane ; lane>=16 -> row r+8 col lane-16
    const int colg  = wave * 16 + mloc;
    const float bv  = bias[colg];
    const int rbase = rtile * 16 + khalf * 8;
#pragma unroll
    for (int r = 0; r < 8; ++r) {
        Xm[(size_t)(rbase + r) * F_MSG + colg] = acc[r] + bv;
    }
}

// ---------------------------------------------------------------------------
// Kernel 2: per-row top-32 of the adjacency + row normalization.
// One wave per row; 8 rows per 256-thread block.
//  - Row (8KB) streamed HBM -> LDS via GLOBAL_LOAD_ASYNC_TO_LDS_B128
//    (ASYNCcnt-tracked, bypasses VGPRs) -- the CDNA5 bulk-stream path.
//  - Selection: each lane caches (max, argmax) over its 16 owned float4
//    chunks (chunk c = lane + 32*jj -> bank-friendly b128 bursts). Per
//    iteration: 5-shuffle wave argmax; only the winning lane marks -inf and
//    rescans its own chunks. No cross-lane LDS visibility needed.
// Ties resolve to the lowest index (matches jax.lax.top_k ordering).
// ---------------------------------------------------------------------------
static constexpr int TK_ROWS_PER_BLOCK = 8;

__global__ __launch_bounds__(256)
void topk_normalize(const float* __restrict__ A,
                    int*   __restrict__ idx_out,
                    float* __restrict__ an_out) {
    __shared__ float buf[TK_ROWS_PER_BLOCK][NNODES];   // 64 KB
    const int lane = threadIdx.x & 31;
    const int w    = threadIdx.x >> 5;
    const long row = (long)blockIdx.x * TK_ROWS_PER_BLOCK + w;

    // ---- async HBM -> LDS stream of this wave's row (16 x 512B bursts).
    // INST_OFFSET is added to BOTH the global and the LDS address (ISA 15.18),
    // so one address pair covers the whole row.
    {
        const unsigned  laddr = (unsigned)(uintptr_t)(&buf[w][0]) + lane * 16;
        const unsigned long long gaddr =
            (unsigned long long)(uintptr_t)(A + row * (long)NNODES) + lane * 16;
        asm volatile("global_load_async_to_lds_b128 %0, %1, off offset:0"    :: "v"(laddr), "v"(gaddr) : "memory");
        asm volatile("global_load_async_to_lds_b128 %0, %1, off offset:512"  :: "v"(laddr), "v"(gaddr) : "memory");
        asm volatile("global_load_async_to_lds_b128 %0, %1, off offset:1024" :: "v"(laddr), "v"(gaddr) : "memory");
        asm volatile("global_load_async_to_lds_b128 %0, %1, off offset:1536" :: "v"(laddr), "v"(gaddr) : "memory");
        asm volatile("global_load_async_to_lds_b128 %0, %1, off offset:2048" :: "v"(laddr), "v"(gaddr) : "memory");
        asm volatile("global_load_async_to_lds_b128 %0, %1, off offset:2560" :: "v"(laddr), "v"(gaddr) : "memory");
        asm volatile("global_load_async_to_lds_b128 %0, %1, off offset:3072" :: "v"(laddr), "v"(gaddr) : "memory");
        asm volatile("global_load_async_to_lds_b128 %0, %1, off offset:3584" :: "v"(laddr), "v"(gaddr) : "memory");
        asm volatile("global_load_async_to_lds_b128 %0, %1, off offset:4096" :: "v"(laddr), "v"(gaddr) : "memory");
        asm volatile("global_load_async_to_lds_b128 %0, %1, off offset:4608" :: "v"(laddr), "v"(gaddr) : "memory");
        asm volatile("global_load_async_to_lds_b128 %0, %1, off offset:5120" :: "v"(laddr), "v"(gaddr) : "memory");
        asm volatile("global_load_async_to_lds_b128 %0, %1, off offset:5632" :: "v"(laddr), "v"(gaddr) : "memory");
        asm volatile("global_load_async_to_lds_b128 %0, %1, off offset:6144" :: "v"(laddr), "v"(gaddr) : "memory");
        asm volatile("global_load_async_to_lds_b128 %0, %1, off offset:6656" :: "v"(laddr), "v"(gaddr) : "memory");
        asm volatile("global_load_async_to_lds_b128 %0, %1, off offset:7168" :: "v"(laddr), "v"(gaddr) : "memory");
        asm volatile("global_load_async_to_lds_b128 %0, %1, off offset:7680" :: "v"(laddr), "v"(gaddr) : "memory");
        asm volatile("s_wait_asynccnt 0x0" ::: "memory");
    }

    // local (max, argmax) over this lane's 16 owned float4 chunks
    const float4* r4 = reinterpret_cast<const float4*>(&buf[w][0]);
    float lmax; int lidx;
    auto rescan = [&]() {
        lmax = -INFINITY; lidx = 0;
#pragma unroll
        for (int jj = 0; jj < NNODES / (32 * 4); ++jj) {
            const int c = lane + 32 * jj;
            const float4 v = r4[c];
            const int base = c * 4;
            if (v.x > lmax) { lmax = v.x; lidx = base;     }
            if (v.y > lmax) { lmax = v.y; lidx = base + 1; }
            if (v.z > lmax) { lmax = v.z; lidx = base + 2; }
            if (v.w > lmax) { lmax = v.w; lidx = base + 3; }
        }
    };
    rescan();

    float myval = 0.0f;
    int   myidx = 0;
    for (int t = 0; t < TOPK; ++t) {
        // wave32 argmax reduction, lowest index wins ties
        float best  = lmax;
        int   besti = lidx;
        for (int off = 16; off > 0; off >>= 1) {
            float ov = __shfl_down(best, off, 32);
            int   oi = __shfl_down(besti, off, 32);
            if (ov > best || (ov == best && oi < besti)) { best = ov; besti = oi; }
        }
        best  = __shfl(best, 0, 32);
        besti = __shfl(besti, 0, 32);
        if (lane == t) { myval = best; myidx = besti; }
        // only the owner lane of the winning slot re-reads LDS
        if (((besti >> 2) & 31) == lane) {
            buf[w][besti] = -INFINITY;
            rescan();
        }
    }

    // normalize: lane t holds t-th largest value
    float s = myval;
    for (int off = 16; off > 0; off >>= 1) s += __shfl_xor(s, off, 32);
    const float an = myval / (s + 1e-12f);

    idx_out[row * TOPK + lane] = myidx;
    an_out [row * TOPK + lane] = an;
}

// ---------------------------------------------------------------------------
// Kernel 3: gather neighbors, BN(eval) + LeakyReLU, weighted reduce.
// One block (128 threads = 1 feature/thread) per node. Xm is only 8.4 MB, so
// the 32 gathers per node (512B coalesced each) hit the 192 MB L2.
// ---------------------------------------------------------------------------
__global__ __launch_bounds__(128)
void aggregate(const float* __restrict__ Xm,
               const int*   __restrict__ idx,
               const float* __restrict__ an,
               const float* __restrict__ gamma,
               const float* __restrict__ beta,
               const float* __restrict__ mean,
               const float* __restrict__ var,
               float* __restrict__ out) {
    __shared__ int   sidx[TOPK];
    __shared__ float san[TOPK];
    const int  m   = threadIdx.x;
    const long row = blockIdx.x;

    if (m < TOPK) {
        sidx[m] = idx[row * TOPK + m];
        san[m]  = an [row * TOPK + m];
    }
    __syncthreads();

    const float g    = gamma[m] * rsqrtf(var[m] + BN_EPS);
    const float bt   = beta[m];
    const float mu   = mean[m];
    const float self = Xm[row * F_MSG + m];

    float acc = 0.0f;
#pragma unroll
    for (int k = 0; k < TOPK; ++k) {
        const float nb = Xm[(size_t)sidx[k] * F_MSG + m];
        float h = (self + nb - mu) * g + bt;
        h = (h >= 0.0f) ? h : LEAKY_SLOPE * h;
        acc = fmaf(san[k], h, acc);
    }
    out[row * F_MSG + m] = acc;
}

// ---------------------------------------------------------------------------
// Launch
// ---------------------------------------------------------------------------
extern "C" void kernel_launch(void* const* d_in, const int* in_sizes, int n_in,
                              void* d_out, int out_size, void* d_ws, size_t ws_size,
                              hipStream_t stream) {
    const float* X     = (const float*)d_in[0];  // [B,N,F_IN]
    const float* A     = (const float*)d_in[1];  // [B,N,N,1]
    const float* W     = (const float*)d_in[2];  // [F_MSG,F_IN]
    const float* bias  = (const float*)d_in[3];  // [F_MSG]
    const float* gamma = (const float*)d_in[4];
    const float* beta  = (const float*)d_in[5];
    const float* mean  = (const float*)d_in[6];
    const float* var   = (const float*)d_in[7];
    float* out = (float*)d_out;                  // [B,N,F_MSG]

    // workspace carve-up
    char* ws = (char*)d_ws;
    float* Xm     = (float*)(ws);                                    // 16384*128 f32
    int*   idxbuf = (int*)  (ws + (size_t)ROWS * F_MSG * 4);         // 16384*32 i32
    float* anbuf  = (float*)(ws + (size_t)ROWS * F_MSG * 4
                                + (size_t)ROWS * TOPK * 4);          // 16384*32 f32

    // 1) projection GEMM (WMMA): 1024 row-tiles x 8 waves
    proj_gemm_wmma<<<ROWS / 16, 256, 0, stream>>>(X, W, bias, Xm);

    // 2) top-k + normalization: 8 rows per block, async HBM->LDS streaming
    topk_normalize<<<ROWS / TK_ROWS_PER_BLOCK, 256, 0, stream>>>(A, idxbuf, anbuf);

    // 3) gather + BN + LeakyReLU + weighted reduce
    aggregate<<<ROWS, F_MSG, 0, stream>>>(Xm, idxbuf, anbuf,
                                          gamma, beta, mean, var, out);
}
